// CF_SMP_79774722556254
// MI455X (gfx1250) — matrix operand order repaired and checked by reading, not compiled, not run on hardware
//
#include <hip/hip_runtime.h>
#include <hip/hip_bf16.h>

// ---------------- problem constants (match reference) ----------------
#define NU   100000
#define NI   50000
#define DD   64
#define NN   150000         // NU + NI   (multiple of 16 -> unguarded WMMA tiles)
#define LL   3
#define HH   128
#define EB_  4000000
#define EU_  1600000
#define EI_  800000
#define BN_EPS 1e-5f

typedef __attribute__((ext_vector_type(16))) _Float16     v16h;
typedef __attribute__((ext_vector_type(8)))  float        v8f;
typedef __attribute__((ext_vector_type(4)))  unsigned int v4u;
typedef __attribute__((ext_vector_type(8)))  int          v8i;
typedef __attribute__((ext_vector_type(4)))  int          v4i;

#if __has_builtin(__builtin_amdgcn_tensor_load_to_lds)
#define USE_TDM 1
#else
#define USE_TDM 0
#endif

// ---------------- small helpers ----------------
__device__ __forceinline__ float u01_hash(unsigned s) {
    s = (s ^ 61u) ^ (s >> 16);
    s *= 9u;
    s ^= s >> 4;
    s *= 0x27d4eb2du;
    s ^= s >> 15;
    return ((float)((s >> 8) + 1u)) * (1.0f / 16777217.0f); // (0,1]
}
__device__ __forceinline__ float gumbel_from(unsigned s) {
    float u = u01_hash(s);
    return -logf(-logf(u + 1e-20f) + 1e-20f);
}

#if USE_TDM
// Issue one TDM 2D tile load: 16 rows x 64 fp32, row stride 64 elems in memory,
// into LDS with 72-DWORD padding after each 64-DWORD row chunk (pitch = 136 f32).
// D# layout per CDNA5 ISA 8.3/8.4 (group0 128b, group1 256b; groups 2/3 unused->0).
__device__ __forceinline__ void tdm_issue_tile(unsigned lds_off, const float* gsrc) {
    unsigned long long ga = (unsigned long long)(size_t)gsrc;
    v4u g0;
    g0.x = 1u;                                            // count=1 valid descriptor
    g0.y = lds_off;                                       // lds_addr [63:32]
    g0.z = (unsigned)ga;                                  // global_addr[31:0]
    g0.w = (unsigned)((ga >> 32) & 0x01ffffffull)         // global_addr[56:32]
           | (2u << 30);                                  // type=2 ("image")
    v8i g1;
    g1[0] = (int)((2u << 16)        // data_size = 4 bytes
                | (1u << 20)        // pad_enable
                | (5u << 22)        // pad_interval: 64 DWORDs
                | (71u << 25));     // pad_amount: 72 DWORDs
    g1[1] = (int)(64u << 16);       // tensor_dim0 = 64 (bits 79:48)
    g1[2] = (int)(16u << 16);       // tensor_dim1 = 16 (bits 111:80)
    g1[3] = (int)(64u << 16);       // tile_dim0   = 64 (bits 127:112)
    g1[4] = 16;                     // tile_dim1   = 16 (bits 143:128), tile_dim2 = 0
    g1[5] = 64;                     // tensor_dim0_stride = 64 (bits 207:160)
    g1[6] = 0;                      // stride hi / tensor_dim1_stride lo
    g1[7] = 0;                      // tensor_dim1_stride hi
    v4i z4 = {0, 0, 0, 0};
#if __clang_major__ >= 23
    v8i z8 = {0, 0, 0, 0, 0, 0, 0, 0};
    __builtin_amdgcn_tensor_load_to_lds(g0, g1, z4, z4, z8, 0);
#else
    __builtin_amdgcn_tensor_load_to_lds(g0, g1, z4, z4, 0);
#endif
}
#endif

// ---------------- init: emb = concat(user,item); accum = emb ----------------
__global__ __launch_bounds__(256) void init_kernel(const float* __restrict__ ue,
                                                   const float* __restrict__ ie,
                                                   float* __restrict__ emb,
                                                   float* __restrict__ accum) {
    long i = (long)blockIdx.x * blockDim.x + threadIdx.x;
    if (i >= (long)NN * DD) return;
    float v = (i < (long)NU * DD) ? ue[i] : ie[i - (long)NU * DD];
    emb[i] = v;
    accum[i] = v;
}

__global__ __launch_bounds__(256) void zero_kernel(float* __restrict__ p, long n) {
    long i = (long)blockIdx.x * blockDim.x + threadIdx.x;
    if (i < n) p[i] = 0.0f;
}

// ---------------- SpMM: wave per edge, float2 per lane ----------------
__global__ __launch_bounds__(256) void spmm_kernel(const int* __restrict__ rows,
                                                   const int* __restrict__ cols,
                                                   const float* __restrict__ vals,
                                                   const float* __restrict__ x,
                                                   float* __restrict__ out,
                                                   int nE) {
    int wave = threadIdx.x >> 5;
    int lane = threadIdx.x & 31;
    long e = (long)blockIdx.x * 8 + wave;
    if (e >= nE) return;
    int r = rows[e];
    int c = cols[e];
    float v = vals[e];
    const float2* xp = (const float2*)(x + (long)c * DD);
    float2 d = xp[lane];
    float* op = out + (long)r * DD + lane * 2;
    atomicAdd(op,     v * d.x);
    atomicAdd(op + 1, v * d.y);
}

// ---------------- gate hidden GEMM: h = feat @ W1 + b1 (WMMA f16 -> f32) ----
// 256 threads = 8 waves; block covers 128 rows (8 strips of 16); each wave owns
// a 16-wide column tile -> 4 x v_wmma_f32_16x16x32_f16 per strip.
// Feat tiles are DMA'd by the Tensor Data Mover into double-buffered LDS,
// strip s+1 in flight while strip s computes (s_wait_tensorcnt 2).
__global__ __launch_bounds__(256) void gemm_h_kernel(const float* __restrict__ cand_ui,
                                                     const float* __restrict__ cand_gcn,
                                                     const float* __restrict__ W1,
                                                     const float* __restrict__ b1,
                                                     float* __restrict__ h,
                                                     float* __restrict__ stats, // [0:128)=sum [128:256)=sumsq
                                                     int nRows) {
    __shared__ _Float16 w1T[HH][HH + 8];        // W1 transposed: [n][k], f16
    __shared__ float    aF32[2][16][HH + 8];    // feat tiles fp32, pitch 136 (TDM pad)
    __shared__ float    colSum[HH], colSq[HH], bias[HH];

    const int t = threadIdx.x;
    const int rowBase = blockIdx.x * 128;

#if USE_TDM
    // Kick off DMA of the first strip's tile; overlaps with W1 staging below.
    const unsigned ldsBuf0 = (unsigned)(unsigned long long)(size_t)&aF32[0][0][0];
    if (t < 32) {
        tdm_issue_tile(ldsBuf0,          cand_ui  + (long)rowBase * DD);
        tdm_issue_tile(ldsBuf0 + DD * 4, cand_gcn + (long)rowBase * DD);
    }
#endif

    // stage W1 transposed as f16 (coalesced global read, scattered LDS write)
    for (int i = t; i < HH * HH; i += 256) {
        int k = i >> 7, n = i & 127;
        w1T[n][k] = (_Float16)W1[i];
    }
    if (t < HH) { bias[t] = b1[t]; colSum[t] = 0.0f; colSq[t] = 0.0f; }
    __syncthreads();

    const int wave = t >> 5, lane = t & 31;
    const int ln15 = lane & 15, lhi = lane >> 4;
    const int n0 = wave * 16;

    // B fragments are strip-invariant: keep in VGPRs.
    // B layout (32x16 f16): lanes 0-15 -> K=kk..kk+15, lanes 16-31 -> K=kk+16..kk+31
    v16h bfrag[4];
#pragma unroll
    for (int kk = 0; kk < 4; ++kk) {
        const _Float16* p = &w1T[n0 + ln15][kk * 32 + 16 * lhi];
#pragma unroll
        for (int j = 0; j < 16; ++j) bfrag[kk][j] = p[j];
    }

    for (int s = 0; s < 8; ++s) {
        const int r0 = rowBase + s * 16;
        if (r0 >= nRows) break;            // nRows % 16 == 0: strips always full
        const int buf = s & 1;

#if USE_TDM
        const int r1 = r0 + 16;
        const bool more = (s < 7) && (r1 < nRows);
        if (t < 32) {
            if (more) {
                unsigned ldsN = (unsigned)(unsigned long long)(size_t)&aF32[buf ^ 1][0][0];
                tdm_issue_tile(ldsN,          cand_ui  + (long)r1 * DD);
                tdm_issue_tile(ldsN + DD * 4, cand_gcn + (long)r1 * DD);
                __builtin_amdgcn_s_wait_tensorcnt(2);  // strip s's pair complete
            } else {
                __builtin_amdgcn_s_wait_tensorcnt(0);
            }
        }
        __syncthreads();                   // publish tile `buf` to all waves
#else
        for (int i = t; i < 16 * HH; i += 256) {
            int m = i >> 7, c = i & 127;
            float v = (c < DD) ? cand_ui[(long)(r0 + m) * DD + c]
                               : cand_gcn[(long)(r0 + m) * DD + (c - DD)];
            aF32[buf][m][c] = v;
        }
        __syncthreads();
#endif

        // A fragments (f32 LDS -> f16), then 4 chained WMMAs.
        // A layout (16x32 f16): lanes<16: K=0..15; lanes>=16: K=16..31
        v16h afrag[4];
#pragma unroll
        for (int kk = 0; kk < 4; ++kk) {
            const float* p = &aF32[buf][ln15][kk * 32 + 16 * lhi];
#pragma unroll
            for (int j = 0; j < 8; ++j) {
                afrag[kk][j]     = (_Float16)p[j];
                afrag[kk][8 + j] = (_Float16)p[8 + j];
            }
        }
        v8f acc = {};
#pragma unroll
        for (int kk = 0; kk < 4; ++kk)
            acc = __builtin_amdgcn_wmma_f32_16x16x32_f16(
                false, afrag[kk], false, bfrag[kk], (short)0, acc, false, false);

        // unguarded store (strips always full) + BN stats
        const int col = n0 + ln15;
        const float bsv = bias[col];
        float* hp = h + (long)(r0 + 8 * lhi) * HH + col;  // C layout: VGPR r -> M=r / r+8
        float lsum = 0.0f, lsq = 0.0f;
#pragma unroll
        for (int r = 0; r < 8; ++r) {
            float v = acc[r] + bsv;
            hp[r * HH] = v;
            lsum += v;
            lsq  += v * v;
        }
        atomicAdd(&colSum[col], lsum);
        atomicAdd(&colSq[col],  lsq);
        __syncthreads();   // all reads of aF32[buf] done before strip s+2's DMA lands
    }
    if (t < HH) {
        atomicAdd(&stats[t],      colSum[t]);
        atomicAdd(&stats[HH + t], colSq[t]);
    }
}

// ---------------- finalize BN stats ----------------
__global__ void finalize_stats_kernel(float* __restrict__ stats, int n) {
    int c = threadIdx.x;
    if (c >= HH) return;
    float inv_n = 1.0f / (float)n;
    float mu = stats[c] * inv_n;
    float var = stats[HH + c] * inv_n - mu * mu;
    stats[2 * HH + c] = mu;
    stats[3 * HH + c] = rsqrtf(var + BN_EPS);
}

// ---------------- gate apply + hard select + accumulate ----------------
// wave per row: normalize/relu, dot with W2, wave-reduce, gumbel argmax, select.
__global__ __launch_bounds__(256) void gate_combine_kernel(const float* __restrict__ h,
                                                           const float* __restrict__ stats,
                                                           const float* __restrict__ gamma,
                                                           const float* __restrict__ beta,
                                                           const float* __restrict__ W2,
                                                           const float* __restrict__ b2,
                                                           const float* __restrict__ cand_ui,
                                                           const float* __restrict__ cand_gcn,
                                                           float* __restrict__ emb,
                                                           float* __restrict__ accum,
                                                           int layer, int nRows) {
    int wave = threadIdx.x >> 5, lane = threadIdx.x & 31;
    int row = blockIdx.x * 8 + wave;
    if (row >= nRows) return;

    float l0 = 0.0f, l1 = 0.0f;
#pragma unroll
    for (int j = 0; j < 4; ++j) {
        int c = lane * 4 + j;
        float v = h[(long)row * HH + c];
        v = (v - stats[2 * HH + c]) * stats[3 * HH + c] * gamma[c] + beta[c];
        v = fmaxf(v, 0.0f);
        l0 += v * W2[c * 2 + 0];
        l1 += v * W2[c * 2 + 1];
    }
#pragma unroll
    for (int off = 16; off > 0; off >>= 1) {
        l0 += __shfl_xor(l0, off, 32);
        l1 += __shfl_xor(l1, off, 32);
    }
    l0 += b2[0];
    l1 += b2[1];

    unsigned seedBase = (unsigned)row * 2u ^ ((unsigned)layer * 2654435761u) ^ 0xC0FFEEu;
    float n0 = gumbel_from(seedBase);
    float n1 = gumbel_from(seedBase + 1u);
    bool pick0 = (l0 + n0) >= (l1 + n1);   // hard gumbel-softmax forward == one-hot argmax

#pragma unroll
    for (int j = 0; j < 2; ++j) {
        long idx = (long)row * DD + lane * 2 + j;
        float v = pick0 ? cand_ui[idx] : cand_gcn[idx];
        emb[idx] = v;
        accum[idx] += v;
    }
}

// ---------------- final: out = accum / (L+1) ----------------
__global__ __launch_bounds__(256) void final_kernel(const float* __restrict__ accum,
                                                    float* __restrict__ out, long n) {
    long i = (long)blockIdx.x * blockDim.x + threadIdx.x;
    if (i < n) out[i] = accum[i] * 0.25f;
}

// ---------------- host driver ----------------
extern "C" void kernel_launch(void* const* d_in, const int* in_sizes, int n_in,
                              void* d_out, int out_size, void* d_ws, size_t ws_size,
                              hipStream_t stream) {
    const float* user_emb  = (const float*)d_in[0];
    const float* item_emb  = (const float*)d_in[1];
    const float* g_vals    = (const float*)d_in[2];
    const float* u_vals    = (const float*)d_in[3];
    const float* i_vals    = (const float*)d_in[4];
    const float* gate_W1   = (const float*)d_in[5];   // (L, 128, 128)
    const float* gate_b1   = (const float*)d_in[6];   // (L, 128)
    const float* gate_gam  = (const float*)d_in[7];   // (L, 128)
    const float* gate_bet  = (const float*)d_in[8];   // (L, 128)
    const float* gate_W2   = (const float*)d_in[9];   // (L, 128, 2)
    const float* gate_b2   = (const float*)d_in[10];  // (L, 2)
    const int*   g_rows    = (const int*)d_in[11];
    const int*   g_cols    = (const int*)d_in[12];
    const int*   u_rows    = (const int*)d_in[13];
    const int*   u_cols    = (const int*)d_in[14];
    const int*   i_rows    = (const int*)d_in[15];
    const int*   i_cols    = (const int*)d_in[16];
    float* out = (float*)d_out;

    const long ND = (long)NN * DD;      // 9.6M floats
    float* ws       = (float*)d_ws;
    float* emb      = ws;               // ND
    float* accum    = emb + ND;         // ND
    float* cand_ui  = accum + ND;       // ND
    float* cand_gcn = cand_ui + ND;     // ND
    float* hbuf     = cand_gcn + ND;    // N*H
    float* stats    = hbuf + (long)NN * HH;  // 512 floats

    const unsigned B = 256;
    init_kernel<<<(unsigned)((ND + B - 1) / B), B, 0, stream>>>(user_emb, item_emb, emb, accum);

    for (int l = 0; l < LL; ++l) {
        zero_kernel<<<(unsigned)((2 * ND + B - 1) / B), B, 0, stream>>>(cand_ui, 2 * ND);
        zero_kernel<<<1, B, 0, stream>>>(stats, 512);

        spmm_kernel<<<(EU_ + 7) / 8, B, 0, stream>>>(u_rows, u_cols, u_vals,
                                                     emb, cand_ui, EU_);
        spmm_kernel<<<(EI_ + 7) / 8, B, 0, stream>>>(i_rows, i_cols, i_vals,
                                                     emb + (long)NU * DD,
                                                     cand_ui + (long)NU * DD, EI_);
        spmm_kernel<<<(EB_ + 7) / 8, B, 0, stream>>>(g_rows, g_cols, g_vals,
                                                     emb, cand_gcn, EB_);

        gemm_h_kernel<<<(NN + 127) / 128, B, 0, stream>>>(cand_ui, cand_gcn,
                                                          gate_W1 + (long)l * 2 * DD * HH,
                                                          gate_b1 + (long)l * HH,
                                                          hbuf, stats, NN);
        finalize_stats_kernel<<<1, 128, 0, stream>>>(stats, NN);

        gate_combine_kernel<<<(NN + 7) / 8, B, 0, stream>>>(hbuf, stats,
                                                            gate_gam + (long)l * HH,
                                                            gate_bet + (long)l * HH,
                                                            gate_W2 + (long)l * HH * 2,
                                                            gate_b2 + (long)l * 2,
                                                            cand_ui, cand_gcn,
                                                            emb, accum, l, NN);
    }
    final_kernel<<<(unsigned)((ND + B - 1) / B), B, 0, stream>>>(accum, out, ND);
}
